// NodeBlockDGL_31705448579494
// MI455X (gfx1250) — matrix operand order repaired and checked by reading, not compiled, not run on hardware
//
#include <hip/hip_runtime.h>

#define N_NODES 10000
#define N_EDGES 160000
#define DIM     512
#define DCAT    1024

typedef __attribute__((ext_vector_type(16))) __bf16 v16bf;
typedef __attribute__((ext_vector_type(8)))  __bf16 v8bf;
typedef __attribute__((ext_vector_type(8)))  float  v8f;

__device__ inline __bf16 f2bf(float f) {
  union { float f; unsigned u; } in; in.f = f;
  unsigned r = in.u + 0x7FFFu + ((in.u >> 16) & 1u);   // round-to-nearest-even
  union { unsigned short s; __bf16 b; } out; out.s = (unsigned short)(r >> 16);
  return out.b;
}

// ---------------------------------------------------------------------------
// Pack W1 [1024x512] and W2 [512x512] (row-major f32, k-major) into bf16
// WMMA B-fragments: Wp[kt][nt][lane][e] with K = kt*32 + (lane/16)*16 + e,
// N = nt*16 + (lane%16). One contiguous 32B load per lane per fragment.
// ---------------------------------------------------------------------------
__global__ __launch_bounds__(256)
void convert_weights_kernel(const float* __restrict__ W1,
                            const float* __restrict__ W2,
                            __bf16* __restrict__ W1p,
                            __bf16* __restrict__ W2p) {
  int t = blockIdx.x * 256 + threadIdx.x;
  const int n1 = 32 * 32 * 32 * 16;           // 524288 elements of W1p
  if (t < n1) {
    int e    = t & 15;
    int lane = (t >> 4) & 31;
    int nt   = (t >> 9) & 31;
    int kt   = t >> 14;                       // 0..31
    int k = kt * 32 + (lane >> 4) * 16 + e;
    int n = nt * 16 + (lane & 15);
    W1p[t] = f2bf(W1[k * DIM + n]);
  } else {
    int u    = t - n1;                        // 0..262143
    int e    = u & 15;
    int lane = (u >> 4) & 31;
    int nt   = (u >> 9) & 31;
    int kt   = u >> 14;                       // 0..15
    int k = kt * 32 + (lane >> 4) * 16 + e;
    int n = nt * 16 + (lane & 15);
    W2p[u] = f2bf(W2[k * DIM + n]);
  }
}

// ---------------------------------------------------------------------------
// segment_sum: one float4 per thread, 4 global_atomic_add_f32 onto agg.
// agg (20 MB) fits in L2 (192 MB) -> atomics resolve at L2 rate.
// ---------------------------------------------------------------------------
__global__ __launch_bounds__(256)
void scatter_add_kernel(const float* __restrict__ efeat,
                        const int* __restrict__ dst,
                        float* __restrict__ agg) {
  long long idx = (long long)blockIdx.x * 256 + threadIdx.x; // float4 index
  int e  = (int)(idx >> 7);       // /128 float4 per edge
  int c4 = (int)(idx & 127);
  const float4 v = ((const float4*)efeat)[idx];
  float* p = agg + (long long)dst[e] * DIM + c4 * 4;
  atomicAdd(p + 0, v.x);
  atomicAdd(p + 1, v.y);
  atomicAdd(p + 2, v.z);
  atomicAdd(p + 3, v.w);
}

// ---------------------------------------------------------------------------
// Fused node MLP: 16-node tile per block; 8 wave32s; each wave owns 4 n-tiles
// (64 cols). bf16 WMMA with f32 accumulate; LayerNorm + residual epilogue.
// ---------------------------------------------------------------------------
__global__ __launch_bounds__(256)
void node_mlp_kernel(const float* __restrict__ nfeat,
                     const float* __restrict__ agg,      // == out (read first)
                     const __bf16* __restrict__ W1p,
                     const __bf16* __restrict__ W2p,
                     const float* __restrict__ b1,
                     const float* __restrict__ b2,
                     const float* __restrict__ lnw,
                     const float* __restrict__ lnb,
                     float* __restrict__ out) {
  // 32 KB region: cat tile (16x1024 bf16) during GEMM1, then h2 (16x512 f32)
  __shared__ __align__(16) unsigned char sa_raw[16 * DCAT * 2];
  __bf16 (*cat)[DCAT] = (__bf16(*)[DCAT])sa_raw;
  float  (*h2)[DIM]   = (float(*)[DIM])sa_raw;
  __shared__ __align__(16) __bf16 hbuf[16][DIM];         // 16 KB
  __shared__ float psum[16][16], psq[16][16], smu[16], srs[16];

  const int t    = threadIdx.x;
  const int lane = t & 31;
  const int wv   = t >> 5;
  const int half = lane >> 4;
  const int m    = lane & 15;
  const int node0 = blockIdx.x * 16;

  // ---- load cat = [agg | nfeat] tile, convert to bf16 ----
  for (int j = 0; j < 64; ++j) {
    int idx = j * 256 + t;                 // 0..16383
    int row = idx >> 10;
    int col = idx & 1023;
    float v = (col < DIM) ? agg[(node0 + row) * DIM + col]
                          : nfeat[(node0 + row) * DIM + (col - DIM)];
    cat[row][col] = f2bf(v);
  }
  __syncthreads();

  // ---- GEMM1: h = cat @ W1  (M16 x K1024 x N64-per-wave) ----
  v8f acc[4] = {};
  #pragma unroll 4
  for (int kt = 0; kt < 32; ++kt) {
    int k0 = kt * 32 + half * 16;
    v8bf a0 = *(const v8bf*)&cat[m][k0];
    v8bf a1 = *(const v8bf*)&cat[m][k0 + 8];
    v16bf a;
    #pragma unroll
    for (int j = 0; j < 8; ++j) { a[j] = a0[j]; a[8 + j] = a1[j]; }
    #pragma unroll
    for (int i = 0; i < 4; ++i) {
      int nt = wv * 4 + i;
      const v16bf b = *(const v16bf*)&W1p[(((kt * 32 + nt) * 32) + lane) * 16];
      acc[i] = __builtin_amdgcn_wmma_f32_16x16x32_bf16(
                   false, a, false, b, (short)0, acc[i], false, false);
    }
  }
  // bias + SiLU -> bf16 h tile
  #pragma unroll
  for (int i = 0; i < 4; ++i) {
    int n = (wv * 4 + i) * 16 + m;
    float bias = b1[n];
    #pragma unroll
    for (int r = 0; r < 8; ++r) {
      int row = r + half * 8;
      float x = acc[i][r] + bias;
      hbuf[row][n] = f2bf(x / (1.f + __expf(-x)));       // SiLU
    }
  }
  __syncthreads();

  // ---- GEMM2: h2 = h @ W2  (K=512) ----
  v8f acc2[4] = {};
  #pragma unroll 4
  for (int kt = 0; kt < 16; ++kt) {
    int k0 = kt * 32 + half * 16;
    v8bf a0 = *(const v8bf*)&hbuf[m][k0];
    v8bf a1 = *(const v8bf*)&hbuf[m][k0 + 8];
    v16bf a;
    #pragma unroll
    for (int j = 0; j < 8; ++j) { a[j] = a0[j]; a[8 + j] = a1[j]; }
    #pragma unroll
    for (int i = 0; i < 4; ++i) {
      int nt = wv * 4 + i;
      const v16bf b = *(const v16bf*)&W2p[(((kt * 32 + nt) * 32) + lane) * 16];
      acc2[i] = __builtin_amdgcn_wmma_f32_16x16x32_bf16(
                    false, a, false, b, (short)0, acc2[i], false, false);
    }
  }
  // bias -> f32 h2 in LDS (aliases cat; all cat reads finished pre-barrier)
  #pragma unroll
  for (int i = 0; i < 4; ++i) {
    int n = (wv * 4 + i) * 16 + m;
    float bias = b2[n];
    #pragma unroll
    for (int r = 0; r < 8; ++r) h2[r + half * 8][n] = acc2[i][r] + bias;
  }
  __syncthreads();

  // ---- LayerNorm (population variance) ----
  {
    int row = t >> 4, sub = t & 15;
    float s = 0.f, sq = 0.f;
    for (int c = sub * 32; c < sub * 32 + 32; ++c) {
      float v = h2[row][c];
      s += v; sq += v * v;
    }
    psum[row][sub] = s; psq[row][sub] = sq;
  }
  __syncthreads();
  if (t < 16) {
    float s = 0.f, sq = 0.f;
    for (int j = 0; j < 16; ++j) { s += psum[t][j]; sq += psq[t][j]; }
    float mu  = s * (1.f / DIM);
    float var = sq * (1.f / DIM) - mu * mu;
    smu[t] = mu;
    srs[t] = rsqrtf(var + 1e-5f);
  }
  __syncthreads();

  // ---- scale/shift + residual, store nfeat_new ----
  for (int j = 0; j < 32; ++j) {
    int idx = j * 256 + t;                 // 0..8191
    int row = idx >> 9;
    int col = idx & 511;
    float v = (h2[row][col] - smu[row]) * srs[row] * lnw[col] + lnb[col]
              + nfeat[(node0 + row) * DIM + col];
    out[(node0 + row) * DIM + col] = v;
  }
}

extern "C" void kernel_launch(void* const* d_in, const int* in_sizes, int n_in,
                              void* d_out, int out_size, void* d_ws, size_t ws_size,
                              hipStream_t stream) {
  const float* efeat = (const float*)d_in[0];
  const float* nfeat = (const float*)d_in[1];
  const int*   dst   = (const int*)d_in[2];
  const float* W1    = (const float*)d_in[3];
  const float* b1    = (const float*)d_in[4];
  const float* W2    = (const float*)d_in[5];
  const float* b2    = (const float*)d_in[6];
  const float* lnw   = (const float*)d_in[7];
  const float* lnb   = (const float*)d_in[8];

  float* out       = (float*)d_out;
  float* out_nfeat = out + (long long)N_EDGES * DIM;   // agg lives here too

  __bf16* W1p = (__bf16*)d_ws;                          // 1 MB
  __bf16* W2p = W1p + 32 * 32 * 32 * 16;                // +0.5 MB

  // 1) passthrough efeat output
  hipMemcpyAsync(out, efeat, (size_t)N_EDGES * DIM * sizeof(float),
                 hipMemcpyDeviceToDevice, stream);
  // 2) zero agg region (overwritten by nfeat_new at the end)
  hipMemsetAsync(out_nfeat, 0, (size_t)N_NODES * DIM * sizeof(float), stream);
  // 3) bf16 weight fragments
  convert_weights_kernel<<<3072, 256, 0, stream>>>(W1, W2, W1p, W2p);
  // 4) segment_sum via f32 atomics
  scatter_add_kernel<<<(N_EDGES * (DIM / 4)) / 256, 256, 0, stream>>>(
      efeat, dst, out_nfeat);
  // 5) fused MLP + LayerNorm + residual
  node_mlp_kernel<<<N_NODES / 16, 256, 0, stream>>>(
      nfeat, out_nfeat, W1p, W2p, b1, b2, lnw, lnb, out_nfeat);
}
